// BaselineFeedforwardNetwork_64854006169779
// MI455X (gfx1250) — matrix operand order repaired and checked by reading, not compiled, not run on hardware
//
#include <hip/hip_runtime.h>

// ---------------------------------------------------------------------------
// Recurrent MLP trajectory forward for MI455X (gfx1250, wave32, WMMA).
//
// Transposed formulation: h^T = W^T @ x^T so the sample dim sits on the
// WMMA B/N side.  Layer-1 D output -> layer-2 B operand needs only a
// relu + f16 pack + one ds_swizzle SWAPX16 lane exchange (no LDS transpose).
// W1^T (bias folded via spare K slot) and W2^T are held as constant A
// fragments in registers per wave.  Layer 3 (64->1) is a per-lane dot +
// SWAPX16 partial-sum exchange, leaving each lane holding the delta of the
// sample it feeds back into next step's x fragment.
//
// NEW this round: features are staged into LDS with the gfx1250 async copy
// engine (global_load_async_to_lds_b128 / s_wait_asynccnt), double-buffered
// in 8-step chunks so the DMA of chunk c+1 overlaps compute of chunk c and
// all global feature traffic is coalesced 16B units (vs 20B@1200B-stride
// scattered loads before).  LDS stride padded to 176B/sample -> per-step
// feature reads are bank-conflict free.
// ---------------------------------------------------------------------------

typedef __attribute__((ext_vector_type(16))) _Float16 v16h;
typedef __attribute__((ext_vector_type(8)))  float    v8f;

constexpr int T_STEPS = 60;
constexpr int FEATN   = 5;
constexpr int HID     = 64;
constexpr int SAMPLES_PER_WAVE  = 16;
constexpr int WAVES_PER_BLOCK   = 8;
constexpr int SAMPLES_PER_BLOCK = SAMPLES_PER_WAVE * WAVES_PER_BLOCK; // 128

constexpr int CH      = 8;                          // time steps per staged chunk
constexpr int NCHUNK  = (T_STEPS + CH - 1) / CH;    // 8 (7 full + 4-step tail)
constexpr int CHBYTES = CH * FEATN * 4;             // 160 B per sample per chunk
constexpr int CHPAD   = CHBYTES + 16;               // 176 B LDS stride (no bank conflicts)
constexpr int UNITS_PER_SAMPLE = CHBYTES / 16;      // 10 x 16B
constexpr int ISSUES  = SAMPLES_PER_WAVE * UNITS_PER_SAMPLE / 32; // 5 async b128 / chunk

union V16HU { v16h v; unsigned u[8]; };
union H2U   { unsigned u; _Float16 h[2]; };

__device__ __forceinline__ unsigned pack2(float a, float b) {
  H2U x; x.h[0] = (_Float16)a; x.h[1] = (_Float16)b; return x.u;
}

// ds_swizzle SWAPX16: lane <-> lane^16 (group-of-32 mode, xor=0x10, and=0x1f)
__device__ __forceinline__ unsigned swap16(unsigned x) {
  return (unsigned)__builtin_amdgcn_ds_swizzle((int)x, 0x401F);
}
__device__ __forceinline__ float swap16f(float x) {
  return __builtin_bit_cast(float,
      __builtin_amdgcn_ds_swizzle(__builtin_bit_cast(int, x), 0x401F));
}

__device__ __forceinline__ float relu(float x) { return x > 0.f ? x : 0.f; }

// Issue one chunk of this wave's features as 5 async b128 copies (all lanes).
// 16B unit u = i*32+lane; sample s = u/10, sub-unit w = u%10.
__device__ __forceinline__ void issue_chunk(const float* __restrict__ featw, int c,
                                            unsigned ldsbase, const float* gclamp,
                                            int lane)
{
  int s = lane / 10;
  int w = lane - 10 * s;
#pragma unroll
  for (int i = 0; i < ISSUES; ++i) {
    const float* g = featw + (size_t)s * (T_STEPS * FEATN) + c * (CH * FEATN) + w * 4;
    if (g > gclamp) g = gclamp;                       // tail-chunk OOB clamp
    const unsigned lds = ldsbase + (unsigned)(s * CHPAD + w * 16);
    asm volatile("global_load_async_to_lds_b128 %0, %1, off"
                 :: "v"(lds), "v"((unsigned long long)(uintptr_t)g)
                 : "memory");
    // advance 32 units: 32 = 3*10 + 2
    w += 2;
    const int cy = (w >= 10) ? 1 : 0;
    w -= 10 * cy;
    s += 3 + cy;
  }
}

__global__ __launch_bounds__(256)
void traj_mlp_wmma(const float* __restrict__ feat,
                   const float* __restrict__ W1, const float* __restrict__ b1,
                   const float* __restrict__ W2, const float* __restrict__ b2,
                   const float* __restrict__ W3, const float* __restrict__ b3,
                   float* __restrict__ out, int n_samples)
{
  // per-wave double-buffered feature staging (45 KB) + per-chunk deltas (4 KB)
  __shared__ __align__(16) unsigned char s_feat[WAVES_PER_BLOCK][2][SAMPLES_PER_WAVE * CHPAD];
  __shared__ float s_dc[WAVES_PER_BLOCK][SAMPLES_PER_WAVE][CH];

  const int lane  = threadIdx.x & 31;
  const int wave  = threadIdx.x >> 5;
  const int n15   = lane & 15;       // sample column within the wave tile
  const int hi    = lane >> 4;       // which K-half of A / B this lane carries
  const int wbase = blockIdx.x * SAMPLES_PER_BLOCK + wave * SAMPLES_PER_WAVE;

  // ---------------- one-time per-wave fragment construction ----------------
  // A1[m] : W1^T tile, rows = hidden 16m..16m+15, K = in_dim (0..4 feat,
  //         5 = delta slot, 6 = bias row, rest zero).  A-layout:
  //         element e of lane -> K = e + 8*((e>>3) + hi).
  v16h A1[4];
#pragma unroll
  for (int m = 0; m < 4; ++m) {
    v16h a = {};
    const int row = 16 * m + n15;
    if (hi == 0) {
#pragma unroll
      for (int e = 0; e < 6; ++e) a[e] = (_Float16)W1[e * HID + row];
      a[6] = (_Float16)b1[row];
    }
    A1[m] = a;
  }

  // A2[m][f] : W2^T, rows = hidden_out 16m.., K = hidden_in 32f + layout slot.
  v16h A2[4][2];
#pragma unroll
  for (int m = 0; m < 4; ++m) {
    const int row = 16 * m + n15;
#pragma unroll
    for (int f = 0; f < 2; ++f) {
      v16h a;
#pragma unroll
      for (int e = 0; e < 16; ++e) {
        const int k = 32 * f + e + 16 * hi;
        a[e] = (_Float16)W2[k * HID + row];
      }
      A2[m][f] = a;
    }
  }

  // Per-lane gathers of b2 / W3 at the hidden indices this lane's D slots hold
  float b2r[4][8], w3r[4][8];
#pragma unroll
  for (int m = 0; m < 4; ++m)
#pragma unroll
    for (int v = 0; v < 8; ++v) {
      const int h = 16 * m + 8 * hi + v;
      b2r[m][v] = b2[h];
      w3r[m][v] = W3[h];
    }
  const float b3s = b3[0];

  const float* featw  = feat + (size_t)wbase * (T_STEPS * FEATN);
  const float* gclamp = feat + (size_t)n_samples * (T_STEPS * FEATN) - 4;
  const unsigned ldsb0 = (unsigned)(uintptr_t)&s_feat[wave][0][0];
  const unsigned ldsb1 = (unsigned)(uintptr_t)&s_feat[wave][1][0];

  issue_chunk(featw, 0, ldsb0, gclamp, lane);   // prime the pipeline

  float delta = 0.0f;
  int buf = 0;

  // ------------------------------ time loop --------------------------------
  for (int c = 0; c < NCHUNK; ++c) {
    if (c + 1 < NCHUNK) {
      issue_chunk(featw, c + 1, buf ? ldsb0 : ldsb1, gclamp, lane);
      asm volatile("s_wait_asynccnt 0x5" ::: "memory");   // chunk c landed
    } else {
      asm volatile("s_wait_asynccnt 0x0" ::: "memory");
    }

    const int t0   = c * CH;
    const int tmax = (T_STEPS - t0) < CH ? (T_STEPS - t0) : CH;
    const float* fbuf = (const float*)&s_feat[wave][buf][0];

    for (int tt = 0; tt < tmax; ++tt) {
      // B1 = x^T : lanes 0-15 carry K=0..6 of their sample's column.
      v16h xb = {};
      if (hi == 0) {
        const float* fl = fbuf + (n15 * (CHPAD / 4) + tt * FEATN);
        xb[0] = (_Float16)fl[0]; xb[1] = (_Float16)fl[1]; xb[2] = (_Float16)fl[2];
        xb[3] = (_Float16)fl[3]; xb[4] = (_Float16)fl[4];
        xb[5] = (_Float16)delta;
        xb[6] = (_Float16)1.0f;                    // bias row
      }

      // ---- layer 1: h1^T = W1^T @ x^T  (bias folded), 4 WMMAs ----
      v8f acc1[4];
#pragma unroll
      for (int m = 0; m < 4; ++m) {
        v8f cz = {};
        acc1[m] = __builtin_amdgcn_wmma_f32_16x16x32_f16(
            false, A1[m], false, xb, (short)0, cz, false, false);
      }

      // relu + pack to f16 pairs (D slots v,v+1 are consecutive hidden idx)
      unsigned h1pk[4][4];
#pragma unroll
      for (int m = 0; m < 4; ++m)
#pragma unroll
        for (int p = 0; p < 4; ++p)
          h1pk[m][p] = pack2(relu(acc1[m][2 * p]), relu(acc1[m][2 * p + 1]));

      // Build B2 fragments: lane n needs K 16*hi + 0..15 of its column; the
      // missing K-half lives in lane n^16 -> one SWAPX16 per packed reg.
      v16h B2f[2];
#pragma unroll
      for (int f = 0; f < 2; ++f) {
        const int mA = 2 * f, mB = 2 * f + 1;
        V16HU r;
#pragma unroll
        for (int p = 0; p < 4; ++p) {
          const unsigned send = hi ? h1pk[mA][p] : h1pk[mB][p];
          const unsigned recv = swap16(send);
          r.u[p]     = hi ? recv        : h1pk[mA][p];
          r.u[4 + p] = hi ? h1pk[mB][p] : recv;
        }
        B2f[f] = r.v;
      }

      // ---- layer 2: h2^T = W2^T @ h1^T, K=64 chained 2 WMMAs x 4 tiles ----
      v8f acc2[4];
#pragma unroll
      for (int m = 0; m < 4; ++m) {
        v8f cz = {};
        cz = __builtin_amdgcn_wmma_f32_16x16x32_f16(
            false, A2[m][0], false, B2f[0], (short)0, cz, false, false);
        acc2[m] = __builtin_amdgcn_wmma_f32_16x16x32_f16(
            false, A2[m][1], false, B2f[1], (short)0, cz, false, false);
      }

      // ---- layer 3: bias + relu + per-lane dot over this lane's 32 hidden
      float p = 0.f;
#pragma unroll
      for (int m = 0; m < 4; ++m)
#pragma unroll
        for (int v = 0; v < 8; ++v) {
          const float h = relu(acc2[m][v] + b2r[m][v]);
          p = fmaf(h, w3r[m][v], p);
        }
      // lanes n and n^16 hold disjoint hidden halves of the SAME sample
      delta = p + swap16f(p) + b3s;

      if (hi == 0) s_dc[wave][n15][tt] = delta;
    }

    // ---- flush this chunk's deltas (32B-contiguous runs per sample) ----
#pragma unroll
    for (int i = 0; i < 4; ++i) {
      const int idx = i * 32 + lane;          // over [16 samples][8 steps]
      const int sl = idx >> 3, ttl = idx & 7;
      if (t0 + ttl < T_STEPS) {
        const int sg = wbase + sl;
        if (sg < n_samples)
          out[(size_t)sg * T_STEPS + t0 + ttl] = s_dc[wave][sl][ttl];
      }
    }
    buf ^= 1;
  }
}

extern "C" void kernel_launch(void* const* d_in, const int* in_sizes, int n_in,
                              void* d_out, int out_size, void* d_ws, size_t ws_size,
                              hipStream_t stream) {
  const float* feat = (const float*)d_in[0];
  const float* W1   = (const float*)d_in[1];
  const float* b1   = (const float*)d_in[2];
  const float* W2   = (const float*)d_in[3];
  const float* b2   = (const float*)d_in[4];
  const float* W3   = (const float*)d_in[5];
  const float* b3   = (const float*)d_in[6];
  float* out        = (float*)d_out;

  const int n = in_sizes[0] / (T_STEPS * FEATN);   // 131072
  const int blocks = (n + SAMPLES_PER_BLOCK - 1) / SAMPLES_PER_BLOCK;
  traj_mlp_wmma<<<blocks, 256, 0, stream>>>(feat, W1, b1, W2, b2, W3, b3, out, n);
}
